// Generator_34677565948841
// MI455X (gfx1250) — matrix operand order repaired and checked by reading, not compiled, not run on hardware
//
#include <hip/hip_runtime.h>
#include <hip/hip_bf16.h>

typedef __attribute__((ext_vector_type(16))) _Float16 v16h;
typedef __attribute__((ext_vector_type(8)))  _Float16 v8h;
typedef __attribute__((ext_vector_type(8)))  float    v8f;

#define SEQ 192
#define SRC 64
#define DIM 256
#define NHD 8
#define DH  32
#define FFD 1024
#define MAXLK 192
#define NPIX 256

// ---------------- WMMA fragment helpers (gfx1250 wave32 layouts) ----------------

__device__ __forceinline__ v8f wmma32(v16h a, v16h b, v8f c) {
  return __builtin_amdgcn_wmma_f32_16x16x32_f16(false, a, false, b, (short)0, c, false, false);
}

// A fragment: 16x32 f16 tile, row-major source, leading dim lda (halfs).
// lane half 0 holds K=0..7 & 16..23, half 1 holds K=8..15 & 24..31 (ISA 7.12.2).
// All call sites guarantee 16B alignment of p and p+16.
__device__ __forceinline__ v16h frag_a(const _Float16* A, int lda) {
  int lane = threadIdx.x & 31;
  int m = lane & 15, h = (lane >> 4) & 1;
  const _Float16* p = A + m * lda + h * 8;
  v8h lo = *(const v8h*)p;
  v8h hi = *(const v8h*)(p + 16);
  return __builtin_shufflevector(lo, hi, 0, 1, 2, 3, 4, 5, 6, 7,
                                 8, 9, 10, 11, 12, 13, 14, 15);
}

// B fragment from TRANSPOSED storage Bt[N,K] row-major (= B column-major):
// b[e] = Bt[n][h*8+e] (e<8), Bt[n][16+h*8+(e-8)] (e>=8) -> matches frag_a K order (2 x b128).
__device__ __forceinline__ v16h frag_bt(const _Float16* Bt, int ld) {
  int lane = threadIdx.x & 31;
  int n = lane & 15, h = (lane >> 4) & 1;
  const _Float16* p = Bt + n * ld + h * 8;
  v8h lo = *(const v8h*)p;
  v8h hi = *(const v8h*)(p + 16);
  return __builtin_shufflevector(lo, hi, 0, 1, 2, 3, 4, 5, 6, 7,
                                 8, 9, 10, 11, 12, 13, 14, 15);
}

// ---------------- threefry2x32 (JAX-exact) ----------------
__device__ __forceinline__ void tf2x32(unsigned k0, unsigned k1, unsigned c0, unsigned c1,
                                       unsigned* o0, unsigned* o1) {
  unsigned ks0 = k0, ks1 = k1, ks2 = k0 ^ k1 ^ 0x1BD11BDAu;
  unsigned x0 = c0 + ks0, x1 = c1 + ks1;
  const int rotA[4] = {13, 15, 26, 6};
  const int rotB[4] = {17, 29, 16, 24};
  unsigned ks[3] = {ks0, ks1, ks2};
#pragma unroll
  for (int i = 0; i < 5; ++i) {
    const int* rot = (i & 1) ? rotB : rotA;
#pragma unroll
    for (int j = 0; j < 4; ++j) {
      x0 += x1;
      int r = rot[j];
      x1 = (x1 << r) | (x1 >> (32 - r));
      x1 ^= x0;
    }
    x0 += ks[(i + 1) % 3];
    x1 += ks[(i + 2) % 3] + (unsigned)(i + 1);
  }
  *o0 = x0; *o1 = x1;
}

// ---------------- small utility kernels ----------------

__global__ void k_zero(float* p, int n) {
  int i = blockIdx.x * blockDim.x + threadIdx.x;
  if (i < n) p[i] = 0.f;
}

// f32 [rows,cols] -> f16 TRANSPOSED: d[(ro + c)*ldT + r] = s[r*cols + c]
__global__ void k_cvt_t(const float* s, _Float16* d, int rows, int cols, int ldT, int ro) {
  int n = rows * cols;
  for (int i = blockIdx.x * blockDim.x + threadIdx.x; i < n; i += gridDim.x * blockDim.x) {
    int r = i / cols, c = i % cols;
    d[(ro + c) * ldT + r] = (_Float16)s[i];
  }
}

// blocks 0..191 -> POS_DEC rows (2D pos enc over 8 x 24 grid), 192..255 -> POS_ENC rows
__global__ void k_pos(float* pdec, float* penc) {
  int b = blockIdx.x, t = threadIdx.x;
  if (b < SEQ) {
    int r = b / 24, c = b % 24;
    float val;
    if (t < 128) {
      int i2 = t >> 1;
      float ang = (float)r * powf(10000.f, -(float)i2 / 64.f);
      val = (t & 1) ? cosf(ang) : sinf(ang);
    } else {
      int tt = t - 128;
      int i2 = tt >> 1;
      float ang = (float)c * powf(10000.f, -(float)i2 / 64.f);
      val = (tt & 1) ? cosf(ang) : sinf(ang);
    }
    pdec[b * DIM + t] = val;
  } else {
    int p = b - SEQ;
    int i2 = t >> 1;
    float ang = (float)p * powf(10000.f, -(float)i2 / 128.f);
    penc[p * DIM + t] = (t & 1) ? cosf(ang) : sinf(ang);
  }
}

__global__ void k_embed(const int* inp, const float* emb, const float* pos, float* xo) {
  int p = blockIdx.x, t = threadIdx.x;
  xo[p * DIM + t] = emb[inp[p] * DIM + t] * 16.0f + pos[p * DIM + t];
}

__global__ void k_build_x(const float* emb, const float* pos, float* x, int istep) {
  int p = blockIdx.x, t = threadIdx.x;
  float v = 0.f;
  if (p <= istep) v = emb[p * DIM + t] * 16.0f + pos[p * DIM + t];
  x[p * DIM + t] = v;
}

// LayerNorm of [rows,256] f32 -> f16 ; one block (256 thr) per row
__global__ void k_ln(const float* X, const float* g, const float* b, _Float16* H) {
  __shared__ float red[256];
  int row = blockIdx.x, t = threadIdx.x;
  float v = X[row * DIM + t];
  red[t] = v; __syncthreads();
  for (int s = 128; s; s >>= 1) { if (t < s) red[t] += red[t + s]; __syncthreads(); }
  float mean = red[0] * (1.f / 256.f);
  __syncthreads();
  float d = v - mean;
  red[t] = d * d; __syncthreads();
  for (int s = 128; s; s >>= 1) { if (t < s) red[t] += red[t + s]; __syncthreads(); }
  float var = red[0] * (1.f / 256.f);
  float y = d * rsqrtf(var + 1e-6f) * g[t] + b[t];
  H[row * DIM + t] = (_Float16)y;
}

// ---------------- WMMA GEMM:  C[MxN] = A[MxK] * B[KxN] (+bias)(relu) ----------------
// B supplied TRANSPOSED (Bt[N,K], leading dim ldbt = K-stride).
// Outputs: C16 (f16 row-major [M,N]), C16t (f16 transposed [N,ldct], vector store),
//          Cacc (f32 residual accumulate, stride N). Any may be null. 8 waves/block.
__global__ void k_gemm(const _Float16* __restrict__ A, int lda,
                       const _Float16* __restrict__ Bt, int ldbt,
                       _Float16* C16, _Float16* C16t, int ldct,
                       float* Cacc, const float* bias,
                       int M, int N, int K, int relu) {
  int wave = blockIdx.x * (blockDim.x >> 5) + (threadIdx.x >> 5);
  int ntn = N >> 4;
  int total = (M >> 4) * ntn;
  if (wave >= total) return;
  int tm = wave / ntn, tn = wave % ntn;
  v8f acc = {};
  for (int k = 0; k < K; k += 32) {
    v16h a = frag_a(A + tm * 16 * lda + k, lda);
    v16h b = frag_bt(Bt + tn * 16 * ldbt + k, ldbt);
    acc = wmma32(a, b, acc);
  }
  int lane = threadIdx.x & 31;
  int n = lane & 15, h = (lane >> 4) & 1;
  int col = tn * 16 + n;
  float vr[8];
#pragma unroll
  for (int r = 0; r < 8; ++r) {
    float v = acc[r];
    if (bias) v += bias[col];
    if (relu) v = v > 0.f ? v : 0.f;
    vr[r] = v;
  }
  if (Cacc) {
#pragma unroll
    for (int r = 0; r < 8; ++r) Cacc[(tm * 16 + h * 8 + r) * N + col] += vr[r];
  }
  if (C16) {
#pragma unroll
    for (int r = 0; r < 8; ++r) C16[(tm * 16 + h * 8 + r) * N + col] = (_Float16)vr[r];
  }
  if (C16t) {
    v8h st;
#pragma unroll
    for (int r = 0; r < 8; ++r) st[r] = (_Float16)vr[r];
    *(v8h*)(C16t + col * ldct + tm * 16 + h * 8) = st;   // rows contiguous -> 16B store
  }
}

// ---------------- fused multi-head attention for one (head, q-tile) ----------------
// 512 threads = 16 waves. scores -> mask -> softmax (LDS) -> prob @ V.
// K in row-major [Lk, ldk]; V supplied TRANSPOSED (Vt[D, ldvt]). maskmode 1 = causal|valid(istep)
__global__ void k_attn(const _Float16* __restrict__ Q, int ldq,
                       const _Float16* __restrict__ Km, int ldk,
                       const _Float16* __restrict__ Vt, int ldvt,
                       _Float16* O, int ldo,
                       int Lq, int Lk, int maskmode, int istep) {
  __shared__ float s_sc[16 * MAXLK];
  __shared__ __align__(16) _Float16 s_p[16 * MAXLK];
  int nqt = Lq >> 4;
  int qt = blockIdx.x % nqt;
  int head = blockIdx.x / nqt;
  int wid = threadIdx.x >> 5;
  int nkt = Lk >> 4;
  int q0 = qt * 16;

  if (wid < nkt) {
    v16h a = frag_a(Q + q0 * ldq + head * DH, ldq);          // 16x32 Q tile
    v16h b = frag_bt(Km + wid * 16 * ldk + head * DH, ldk);  // (K tile)^T, contiguous rows
    v8f acc = {};
    acc = wmma32(a, b, acc);
    int lane = threadIdx.x & 31;
    int n = lane & 15, h = (lane >> 4) & 1;
#pragma unroll
    for (int r = 0; r < 8; ++r) {
      int lr = h * 8 + r;
      int qpos = q0 + lr;
      int kpos = wid * 16 + n;
      float s = acc[r] * 0.17677669529663689f;               // 1/sqrt(32)
      if (maskmode == 1 && (kpos > qpos || kpos > istep)) s = -1e9f;
      s_sc[lr * MAXLK + kpos] = s;
    }
  }
  __syncthreads();

  if (threadIdx.x < 16) {
    int r = threadIdx.x;
    float mx = -3.4e38f;
    for (int k2 = 0; k2 < Lk; ++k2) mx = fmaxf(mx, s_sc[r * MAXLK + k2]);
    float sum = 0.f;
    for (int k2 = 0; k2 < Lk; ++k2) {
      float e = expf(s_sc[r * MAXLK + k2] - mx);
      s_sc[r * MAXLK + k2] = e;
      sum += e;
    }
    float inv = 1.0f / sum;
    for (int k2 = 0; k2 < Lk; ++k2)
      s_p[r * MAXLK + k2] = (_Float16)(s_sc[r * MAXLK + k2] * inv);
  }
  __syncthreads();

  if (wid < 2) {                                             // out tile [16 x 16], 2 per head
    v8f acc = {};
    for (int k2 = 0; k2 < Lk; k2 += 32) {
      v16h a = frag_a(s_p + k2, MAXLK);
      v16h b = frag_bt(Vt + (head * DH + wid * 16) * ldvt + k2, ldvt);
      acc = wmma32(a, b, acc);
    }
    int lane = threadIdx.x & 31;
    int n = lane & 15, h = (lane >> 4) & 1;
#pragma unroll
    for (int r = 0; r < 8; ++r)
      O[(q0 + h * 8 + r) * ldo + head * DH + wid * 16 + n] = (_Float16)acc[r];
  }
}

// ---------------- final LN(row i) + logits + threefry/gumbel sampling ----------------
__global__ void k_finalize(const float* x, const float* dg, const float* db,
                           const float* w_out, const float* b_out,
                           const float* t_emb, float* emb_buf,
                           float* logits_out, float* tok_out, int istep) {
  __shared__ float red[256];
  __shared__ float lnrow[256];
  __shared__ int   idxs[256];
  int t = threadIdx.x;
  float v = x[istep * DIM + t];
  red[t] = v; __syncthreads();
  for (int s = 128; s; s >>= 1) { if (t < s) red[t] += red[t + s]; __syncthreads(); }
  float mean = red[0] * (1.f / 256.f);
  __syncthreads();
  float d = v - mean;
  red[t] = d * d; __syncthreads();
  for (int s = 128; s; s >>= 1) { if (t < s) red[t] += red[t + s]; __syncthreads(); }
  float var = red[0] * (1.f / 256.f);
  lnrow[t] = d * rsqrtf(var + 1e-6f) * dg[t] + db[t];
  __syncthreads();

  float acc = b_out[t];
  for (int k = 0; k < DIM; ++k) acc = fmaf(lnrow[k], w_out[k * NPIX + t], acc);
  logits_out[istep * NPIX + t] = acc;

  // subkey i of jax.random.split(key(1), 192): keys = threefry(k=(0,1), iota(384)).reshape(192,2)
  unsigned ka, kb, a0, a1;
  unsigned j0 = 2u * (unsigned)istep, j1 = j0 + 1u;
  if (j0 < 192u) { tf2x32(0u, 1u, j0, 192u + j0, &a0, &a1); ka = a0; }
  else           { tf2x32(0u, 1u, j0 - 192u, j0, &a0, &a1); ka = a1; }
  if (j1 < 192u) { tf2x32(0u, 1u, j1, 192u + j1, &a0, &a1); kb = a0; }
  else           { tf2x32(0u, 1u, j1 - 192u, j1, &a0, &a1); kb = a1; }

  // random_bits(key, 32, (256,)): counts split halves
  unsigned y0, y1, bits;
  if (t < 128) { tf2x32(ka, kb, (unsigned)t, (unsigned)(128 + t), &y0, &y1); bits = y0; }
  else         { tf2x32(ka, kb, (unsigned)(t - 128), (unsigned)t, &y0, &y1); bits = y1; }
  float u = __uint_as_float((bits >> 9) | 0x3f800000u) - 1.0f;
  u = fmaxf(u, 1.17549435e-38f);
  float gum = -logf(-logf(u));

  red[t] = acc + gum; idxs[t] = t; __syncthreads();
  for (int s = 128; s; s >>= 1) {
    if (t < s) {
      float o = red[t + s];
      if (o > red[t] || (o == red[t] && idxs[t + s] < idxs[t])) { red[t] = o; idxs[t] = idxs[t + s]; }
    }
    __syncthreads();
  }
  int tok = idxs[0];
  if (istep + 1 < SEQ) emb_buf[(istep + 1) * DIM + t] = t_emb[tok * DIM + t];
  if (t == 0) tok_out[istep] = (float)tok;
}

// ---------------- host orchestration ----------------

extern "C" void kernel_launch(void* const* d_in, const int* in_sizes, int n_in,
                              void* d_out, int out_size, void* d_ws, size_t ws_size,
                              hipStream_t stream) {
  (void)in_sizes; (void)n_in; (void)out_size; (void)ws_size;
  auto f = [&](int i) { return (const float*)d_in[i]; };
  const int* inp = (const int*)d_in[0];

  // sorted-key pytree flatten: 0 input_data, 1 b_out, 2.. dec_layers[0] {cross wk,wo,wq,wv;
  // ffn b1,b2,w1,w2; ln1_b,ln1_g,ln2_b,ln2_g,ln3_b,ln3_g; self wk,wo,wq,wv}, 20.. dec_layers[1],
  // 38 dec_ln_b, 39 dec_ln_g, 40.. enc_layers[0] {attn wk,wo,wq,wv; ffn b1,b2,w1,w2;
  // ln1_b,ln1_g,ln2_b,ln2_g}, 52.. enc_layers[1], 64 enc_ln_b, 65 enc_ln_g,
  // 66 src_emb, 67 t_emb, 68 w_out
  char* wsb = (char*)d_ws;
  size_t off = 0;
  auto alloc = [&](size_t bytes) -> void* {
    void* p = wsb + off;
    off = (off + bytes + 255) & ~(size_t)255;
    return p;
  };
  // f16 weights, all stored TRANSPOSED [N,K]
  _Float16 *dqkT[2], *dvT[2], *dwoT[2], *dcqT[2], *dckT[2], *dcvT[2], *dwocT[2], *dw1T[2], *dw2T[2];
  _Float16 *eqkT[2], *evT[2], *ewoT[2], *ew1T[2], *ew2T[2];
  for (int l = 0; l < 2; ++l) {
    dqkT[l]  = (_Float16*)alloc(512 * 256 * 2);   // wq^T | wk^T
    dvT[l]   = (_Float16*)alloc(256 * 256 * 2);
    dwoT[l]  = (_Float16*)alloc(256 * 256 * 2);
    dcqT[l]  = (_Float16*)alloc(256 * 256 * 2);
    dckT[l]  = (_Float16*)alloc(256 * 256 * 2);
    dcvT[l]  = (_Float16*)alloc(256 * 256 * 2);
    dwocT[l] = (_Float16*)alloc(256 * 256 * 2);
    dw1T[l]  = (_Float16*)alloc(1024 * 256 * 2);
    dw2T[l]  = (_Float16*)alloc(256 * 1024 * 2);
    eqkT[l]  = (_Float16*)alloc(512 * 256 * 2);
    evT[l]   = (_Float16*)alloc(256 * 256 * 2);
    ewoT[l]  = (_Float16*)alloc(256 * 256 * 2);
    ew1T[l]  = (_Float16*)alloc(1024 * 256 * 2);
    ew2T[l]  = (_Float16*)alloc(256 * 1024 * 2);
  }
  // activations / state
  float* pos_dec = (float*)alloc(SEQ * DIM * 4);
  float* pos_enc = (float*)alloc(SRC * DIM * 4);
  float* xe      = (float*)alloc(SRC * DIM * 4);
  _Float16* h16e  = (_Float16*)alloc(SRC * DIM * 2);
  _Float16* eqk16 = (_Float16*)alloc(SRC * 512 * 2);
  _Float16* evT16 = (_Float16*)alloc(256 * SRC * 2);   // V^T [D, SRC]
  _Float16* eo16  = (_Float16*)alloc(SRC * DIM * 2);
  _Float16* eg16  = (_Float16*)alloc(SRC * FFD * 2);
  _Float16* enc16 = (_Float16*)alloc(SRC * DIM * 2);
  _Float16 *kc16[2], *vcT16[2];
  for (int l = 0; l < 2; ++l) {
    kc16[l]  = (_Float16*)alloc(SRC * DIM * 2);        // [SRC, D] row-major
    vcT16[l] = (_Float16*)alloc(256 * SRC * 2);        // V^T [D, SRC]
  }
  float* emb = (float*)alloc(SEQ * DIM * 4);
  float* x   = (float*)alloc(SEQ * DIM * 4);
  _Float16* h16  = (_Float16*)alloc(SEQ * DIM * 2);
  _Float16* qk16 = (_Float16*)alloc(SEQ * 512 * 2);    // self Q|K (or cross Q, ld 256)
  _Float16* vT16 = (_Float16*)alloc(256 * SEQ * 2);    // self V^T [D, SEQ]
  _Float16* o16  = (_Float16*)alloc(SEQ * DIM * 2);
  _Float16* g16  = (_Float16*)alloc(SEQ * FFD * 2);

  float* logits_out = (float*)d_out;
  float* tok_out    = logits_out + SEQ * NPIX;

  auto cvt = [&](const float* s, _Float16* d2, int r, int c, int ldT, int ro) {
    int n = r * c;
    int blocks = (n + 255) / 256;
    if (blocks > 1024) blocks = 1024;
    k_cvt_t<<<blocks, 256, 0, stream>>>(s, d2, r, c, ldT, ro);
  };
  auto gemm = [&](const _Float16* A, int lda, const _Float16* Bt, int ldbt,
                  _Float16* C16, _Float16* C16t, int ldct,
                  float* Cacc, const float* bias, int M, int N, int K, int relu) {
    int tiles = (M / 16) * (N / 16);
    int blocks = (tiles + 7) / 8;
    k_gemm<<<blocks, 256, 0, stream>>>(A, lda, Bt, ldbt, C16, C16t, ldct, Cacc, bias,
                                       M, N, K, relu);
  };

  // ---- weight prep (every call; deterministic). ldT = original rows (K dim). ----
  for (int l = 0; l < 2; ++l) {
    int bi = 2 + 18 * l;
    cvt(f(bi + 16), dqkT[l], 256, 256, 256, 0);    // self.wq -> rows 0..255
    cvt(f(bi + 14), dqkT[l], 256, 256, 256, 256);  // self.wk -> rows 256..511
    cvt(f(bi + 17), dvT[l],  256, 256, 256, 0);    // self.wv
    cvt(f(bi + 15), dwoT[l], 256, 256, 256, 0);    // self.wo
    cvt(f(bi + 2),  dcqT[l], 256, 256, 256, 0);    // cross.wq
    cvt(f(bi + 0),  dckT[l], 256, 256, 256, 0);    // cross.wk
    cvt(f(bi + 3),  dcvT[l], 256, 256, 256, 0);    // cross.wv
    cvt(f(bi + 1),  dwocT[l], 256, 256, 256, 0);   // cross.wo
    cvt(f(bi + 6),  dw1T[l], 256, 1024, 256, 0);   // ffn.w1 [256,1024] -> [1024,256]
    cvt(f(bi + 7),  dw2T[l], 1024, 256, 1024, 0);  // ffn.w2 [1024,256] -> [256,1024]
    int ei = 40 + 12 * l;
    cvt(f(ei + 2), eqkT[l], 256, 256, 256, 0);
    cvt(f(ei + 0), eqkT[l], 256, 256, 256, 256);
    cvt(f(ei + 3), evT[l],  256, 256, 256, 0);
    cvt(f(ei + 1), ewoT[l], 256, 256, 256, 0);
    cvt(f(ei + 6), ew1T[l], 256, 1024, 256, 0);
    cvt(f(ei + 7), ew2T[l], 1024, 256, 1024, 0);
  }
  k_pos<<<256, 256, 0, stream>>>(pos_dec, pos_enc);
  k_zero<<<(SEQ * DIM + 255) / 256, 256, 0, stream>>>(emb, SEQ * DIM);

  // ---- encoder (once) ----
  k_embed<<<SRC, 256, 0, stream>>>(inp, f(66), pos_enc, xe);
  for (int l = 0; l < 2; ++l) {
    int ei = 40 + 12 * l;
    k_ln<<<SRC, 256, 0, stream>>>(xe, f(ei + 9), f(ei + 8), h16e);
    gemm(h16e, 256, eqkT[l], 256, eqk16, nullptr, 0, nullptr, nullptr, SRC, 512, 256, 0);
    gemm(h16e, 256, evT[l], 256, nullptr, evT16, SRC, nullptr, nullptr, SRC, 256, 256, 0);
    k_attn<<<NHD * (SRC / 16), 512, 0, stream>>>(eqk16, 512, eqk16 + 256, 512,
                                                 evT16, SRC, eo16, 256, SRC, SRC, 0, 0);
    gemm(eo16, 256, ewoT[l], 256, nullptr, nullptr, 0, xe, nullptr, SRC, 256, 256, 0);
    k_ln<<<SRC, 256, 0, stream>>>(xe, f(ei + 11), f(ei + 10), h16e);
    gemm(h16e, 256, ew1T[l], 256, eg16, nullptr, 0, nullptr, f(ei + 4), SRC, 1024, 256, 1);
    gemm(eg16, 1024, ew2T[l], 1024, nullptr, nullptr, 0, xe, f(ei + 5), SRC, 256, 1024, 0);
  }
  k_ln<<<SRC, 256, 0, stream>>>(xe, f(65), f(64), enc16);
  // precompute cross K/V per decoder layer (invariant across steps)
  for (int l = 0; l < 2; ++l) {
    gemm(enc16, 256, dckT[l], 256, kc16[l], nullptr, 0, nullptr, nullptr, SRC, 256, 256, 0);
    gemm(enc16, 256, dcvT[l], 256, nullptr, vcT16[l], SRC, nullptr, nullptr, SRC, 256, 256, 0);
  }

  // ---- autoregressive decode: 192 full decoder passes ----
  for (int i = 0; i < SEQ; ++i) {
    k_build_x<<<SEQ, 256, 0, stream>>>(emb, pos_dec, x, i);
    for (int l = 0; l < 2; ++l) {
      int bi = 2 + 18 * l;
      // self-attention
      k_ln<<<SEQ, 256, 0, stream>>>(x, f(bi + 9), f(bi + 8), h16);
      gemm(h16, 256, dqkT[l], 256, qk16, nullptr, 0, nullptr, nullptr, SEQ, 512, 256, 0);
      gemm(h16, 256, dvT[l], 256, nullptr, vT16, SEQ, nullptr, nullptr, SEQ, 256, 256, 0);
      k_attn<<<NHD * (SEQ / 16), 512, 0, stream>>>(qk16, 512, qk16 + 256, 512,
                                                   vT16, SEQ, o16, 256, SEQ, SEQ, 1, i);
      gemm(o16, 256, dwoT[l], 256, nullptr, nullptr, 0, x, nullptr, SEQ, 256, 256, 0);
      // cross-attention
      k_ln<<<SEQ, 256, 0, stream>>>(x, f(bi + 11), f(bi + 10), h16);
      gemm(h16, 256, dcqT[l], 256, qk16, nullptr, 0, nullptr, nullptr, SEQ, 256, 256, 0);
      k_attn<<<NHD * (SEQ / 16), 512, 0, stream>>>(qk16, 256, kc16[l], 256,
                                                   vcT16[l], SRC, o16, 256, SEQ, SRC, 0, 0);
      gemm(o16, 256, dwocT[l], 256, nullptr, nullptr, 0, x, nullptr, SEQ, 256, 256, 0);
      // FFN
      k_ln<<<SEQ, 256, 0, stream>>>(x, f(bi + 13), f(bi + 12), h16);
      gemm(h16, 256, dw1T[l], 256, g16, nullptr, 0, nullptr, f(bi + 4), SEQ, 1024, 256, 1);
      gemm(g16, 1024, dw2T[l], 1024, nullptr, nullptr, 0, x, f(bi + 5), SEQ, 256, 1024, 0);
    }
    k_finalize<<<1, 256, 0, stream>>>(x, f(39), f(38), f(68), f(1), f(67),
                                      emb, logits_out, tok_out, i);
  }
}
